// TopKSAE_58265526338069
// MI455X (gfx1250) — matrix operand order repaired and checked by reading, not compiled, not run on hardware
//
#include <hip/hip_runtime.h>

typedef __attribute__((ext_vector_type(4)))  __bf16 bf16x4;
typedef __attribute__((ext_vector_type(8)))  __bf16 bf16x8;
typedef __attribute__((ext_vector_type(16))) __bf16 bf16x16;
typedef __attribute__((ext_vector_type(8)))  float  f32x8;
typedef __attribute__((ext_vector_type(4)))  int    i32x4;

typedef __attribute__((address_space(1))) void  as1_void;
typedef __attribute__((address_space(3))) void  as3_void;
typedef __attribute__((address_space(1))) i32x4 as1_i32x4;
typedef __attribute__((address_space(3))) i32x4 as3_i32x4;

constexpr int Bb = 4096;
constexpr int Dd = 2048;
constexpr int Nn = 32768;
constexpr int Kk = 64;

constexpr int BM = 128, BN = 128, BK = 32, LDT = 40; // LDT = padded LDS row (elements)

// ---- CDNA5 async global->LDS path (guarded; falls back to reg-staged copy) ----------
#if defined(__has_builtin)
#if __has_builtin(__builtin_amdgcn_global_load_async_to_lds_b128)
#define HAVE_ASYNC_LDS 1
#endif
#endif

__device__ __forceinline__ void lds_copy16(const __bf16* g, __bf16* l) {
#ifdef HAVE_ASYNC_LDS
    __builtin_amdgcn_global_load_async_to_lds_b128(
        (as1_i32x4*)(as1_void*)(__bf16*)g,   // global src (v4i, AS1)
        (as3_i32x4*)(as3_void*)l,            // LDS dst   (v4i, AS3)
        /*offset=*/0, /*cpol=*/0);
#else
    *(bf16x8*)l = *(const bf16x8*)g;
#endif
}

template <int NCNT>
__device__ __forceinline__ void wait_async() {
#ifdef HAVE_ASYNC_LDS
#if __has_builtin(__builtin_amdgcn_s_wait_asynccnt)
    __builtin_amdgcn_s_wait_asynccnt(NCNT);
#else
    if (NCNT == 0) asm volatile("s_wait_asynccnt 0x0" ::: "memory");
    else           asm volatile("s_wait_asynccnt 0x4" ::: "memory");
#endif
#endif
}

__device__ __forceinline__ __bf16 f2bf(float f) {
    // round-to-nearest-even fp32 -> bf16
    unsigned u = __builtin_bit_cast(unsigned, f);
    unsigned r = u + 0x7FFFu + ((u >> 16) & 1u);
    unsigned short s = (unsigned short)(r >> 16);
    return __builtin_bit_cast(__bf16, s);
}

// ---------------- prep: x_cent = x - b_dec, cast to bf16; zero the loss slot ---------
__global__ __launch_bounds__(256)
void center_cast_kernel(const float* __restrict__ x, const float* __restrict__ b_dec,
                        __bf16* __restrict__ xc, float* __restrict__ loss) {
    const int i = blockIdx.x * 256 + threadIdx.x;           // covers B*D
    const float v = x[i] - b_dec[i & (Dd - 1)];
    xc[i] = f2bf(v);
    if (i == 0) *loss = 0.0f;
}

// ---------------- prep: W_enc -> bf16 (vectorized x4) --------------------------------
__global__ __launch_bounds__(256)
void castw_kernel(const float* __restrict__ W, __bf16* __restrict__ Wb) {
    const size_t i4 = (size_t)blockIdx.x * 256 + threadIdx.x;  // covers N*D/4
    const float4 w = ((const float4*)W)[i4];
    bf16x4 o;
    o[0] = f2bf(w.x); o[1] = f2bf(w.y); o[2] = f2bf(w.z); o[3] = f2bf(w.w);
    ((bf16x4*)Wb)[i4] = o;
}

// ---------------- encoder: z_pre = relu(Xc @ Wb^T + b_enc) ---------------------------
// Double-buffered LDS tiles fed by async global->LDS; v_wmma_f32_16x16x32_bf16 compute.
__global__ __launch_bounds__(256)
void encode_kernel(const __bf16* __restrict__ Xc, const __bf16* __restrict__ Wb,
                   const float* __restrict__ b_enc, float* __restrict__ z_pre) {
    __shared__ __bf16 As[2][BM * LDT];   // 2 x 10 KB
    __shared__ __bf16 Bs[2][BN * LDT];   // 2 x 10 KB

    const int tid  = threadIdx.x;
    const int lane = tid & 31;
    const int wid  = tid >> 5;        // 0..7
    const int wm   = wid >> 2;        // 0..1  (M direction, 64 rows each)
    const int wn   = wid & 3;         // 0..3  (N direction, 32 cols each)
    const int m0   = blockIdx.y * BM;
    const int n0   = blockIdx.x * BN;
    const int l15  = lane & 15;

    f32x8 acc[4][2] = {};

    // global->LDS mapping: thread t moves 16 bf16 (2 x b128) of row t>>1, col (t&1)*16
    const int lr = tid >> 1;
    const int lc = (tid & 1) << 4;
    const __bf16* Xg = Xc + (size_t)(m0 + lr) * Dd + lc;
    const __bf16* Wg = Wb + (size_t)(n0 + lr) * Dd + lc;
    const int lds_off = lr * LDT + lc;

    // per-lane K bases per ISA fragment layouts (16-bit A 16x32, B 32x16)
    const int ak = (lane < 16) ? 0 : 8;
    const int bk = (lane < 16) ? 0 : 8;

    auto issue_tile = [&](int it, int buf) {
        const int k0 = it * BK;
        lds_copy16(Xg + k0,     &As[buf][lds_off]);
        lds_copy16(Xg + k0 + 8, &As[buf][lds_off + 8]);
        lds_copy16(Wg + k0,     &Bs[buf][lds_off]);
        lds_copy16(Wg + k0 + 8, &Bs[buf][lds_off + 8]);
    };

    constexpr int NIT = Dd / BK;      // 64
    issue_tile(0, 0);

    for (int it = 0; it < NIT; ++it) {
        const int cur = it & 1;
        if (it + 1 < NIT) {           // prefetch next tile into the other buffer
            issue_tile(it + 1, cur ^ 1);
            wait_async<4>();          // in-order: oldest 4 (current tile) are done
        } else {
            wait_async<0>();
        }
        __syncthreads();              // every thread waited -> whole tile visible

        bf16x16 afrag[4], bfrag[2];
#pragma unroll
        for (int mi = 0; mi < 4; ++mi) {
            const int row = wm * 64 + mi * 16 + l15;
            union { bf16x16 v; bf16x8 h[2]; } u;
            u.h[0] = *(const bf16x8*)&As[cur][row * LDT + ak];       // K = ak..ak+7
            u.h[1] = *(const bf16x8*)&As[cur][row * LDT + ak + 16];  // K = ak+16..+23
            afrag[mi] = u.v;
        }
#pragma unroll
        for (int ni = 0; ni < 2; ++ni) {
            const int col = wn * 32 + ni * 16 + l15;
            union { bf16x16 v; bf16x8 h[2]; } u;
            u.h[0] = *(const bf16x8*)&Bs[cur][col * LDT + bk];       // K = bk..bk+7
            u.h[1] = *(const bf16x8*)&Bs[cur][col * LDT + bk + 16];  // K = bk+16..+23
            bfrag[ni] = u.v;
        }
#pragma unroll
        for (int mi = 0; mi < 4; ++mi)
#pragma unroll
            for (int ni = 0; ni < 2; ++ni)
                acc[mi][ni] = __builtin_amdgcn_wmma_f32_16x16x32_bf16(
                    false, afrag[mi], false, bfrag[ni],
                    (short)0, acc[mi][ni], false, false);
        __syncthreads();              // done reading 'cur' before it is refilled
    }

    // epilogue: C layout -> +bias, relu, store
    const int roff = (lane < 16) ? 0 : 8;
#pragma unroll
    for (int ni = 0; ni < 2; ++ni) {
        const int col = n0 + wn * 32 + ni * 16 + l15;
        const float be = b_enc[col];
#pragma unroll
        for (int mi = 0; mi < 4; ++mi) {
#pragma unroll
            for (int v = 0; v < 8; ++v) {
                const int row = m0 + wm * 64 + mi * 16 + v + roff;
                float r = acc[mi][ni][v] + be;
                z_pre[(size_t)row * Nn + col] = r > 0.f ? r : 0.f;
            }
        }
    }
}

// ---------------- top-K: exact radix select on float bits (row cached in LDS) --------
__global__ __launch_bounds__(256)
void topk_kernel(const float* __restrict__ z_pre, float* __restrict__ z_topk,
                 float* __restrict__ tk_idx_f, int* __restrict__ ws_idx,
                 float* __restrict__ ws_val) {
    extern __shared__ unsigned smem[];
    unsigned* row  = smem;        // N entries (128 KB)
    unsigned* hist = smem + Nn;   // 256 bins
    __shared__ unsigned s_pref, s_need, s_sel, s_eq;

    const int b = blockIdx.x;
    const int t = threadIdx.x;
    const float* zr = z_pre + (size_t)b * Nn;
    for (int i = t; i < Nn; i += 256) row[i] = __builtin_bit_cast(unsigned, zr[i]);
    if (t == 0) { s_pref = 0u; s_need = (unsigned)Kk; s_sel = 0u; s_eq = 0u; }
    __syncthreads();

    // z_pre >= 0 after relu: uint ordering == float ordering
    for (int shift = 24; shift >= 0; shift -= 8) {
        hist[t] = 0u;
        __syncthreads();
        const unsigned pref = s_pref;
        for (int i = t; i < Nn; i += 256) {
            const unsigned u = row[i];
            const bool match = (shift == 24) || ((u >> (shift + 8)) == pref);
            if (match) atomicAdd(&hist[(u >> shift) & 0xFFu], 1u);
        }
        __syncthreads();
        if (t == 0) {
            unsigned need = s_need, cum = 0; int bin = 0;
            for (int j = 255; j >= 0; --j) {
                const unsigned h = hist[j];
                if (cum + h >= need) { bin = j; break; }
                cum += h;
            }
            s_need = need - cum;                 // ties of final threshold to take
            s_pref = (pref << 8) | (unsigned)bin;
        }
        __syncthreads();
    }

    const unsigned T = s_pref;   // bit pattern of K-th largest
    const unsigned r = s_need;   // #elements equal to T to include
    float* zt = z_topk + (size_t)b * Nn;
    for (int i = t; i < Nn; i += 256) {
        const unsigned u = row[i];
        float out = 0.f;
        if (u > T) {
            out = __builtin_bit_cast(float, u);
            const unsigned slot = atomicAdd(&s_sel, 1u);
            ws_idx[b * Kk + slot] = i;
            ws_val[b * Kk + slot] = out;
        } else if (u == T) {
            const unsigned e = atomicAdd(&s_eq, 1u);
            if (e < r) {
                out = __builtin_bit_cast(float, u);
                const unsigned slot = atomicAdd(&s_sel, 1u);
                ws_idx[b * Kk + slot] = i;
                ws_val[b * Kk + slot] = out;
            }
        }
        zt[i] = out;
    }
    __syncthreads();
    if (t < Kk) tk_idx_f[b * Kk + t] = (float)ws_idx[b * Kk + t];
}

// ---------------- decoder: x_hat = scatter(topk) @ W_dec^T + b_dec; fused MSE --------
__global__ __launch_bounds__(256)
void decode_kernel(const float* __restrict__ x, const float* __restrict__ W_dec,
                   const float* __restrict__ b_dec, const int* __restrict__ ws_idx,
                   const float* __restrict__ ws_val, float* __restrict__ x_hat,
                   float* __restrict__ loss) {
    __shared__ int   sidx[Kk];
    __shared__ float sval[Kk];
    __shared__ float red[256];
    const int b = blockIdx.x;
    const int t = threadIdx.x;
    if (t < Kk) { sidx[t] = ws_idx[b * Kk + t]; sval[t] = ws_val[b * Kk + t]; }
    __syncthreads();

    float acc[8];
#pragma unroll
    for (int j = 0; j < 8; ++j) acc[j] = b_dec[t + j * 256];
    for (int k = 0; k < Kk; ++k) {
        const int   n = sidx[k];
        const float v = sval[k];
#pragma unroll
        for (int j = 0; j < 8; ++j)
            acc[j] += v * W_dec[(size_t)(t + j * 256) * Nn + n];  // L2-resident gather
    }
    float sq = 0.f;
#pragma unroll
    for (int j = 0; j < 8; ++j) {
        const int d = t + j * 256;
        const float e = acc[j] - x[(size_t)b * Dd + d];
        sq += e * e;
        x_hat[(size_t)b * Dd + d] = acc[j];
    }
    red[t] = sq;
    __syncthreads();
    for (int s = 128; s > 0; s >>= 1) {
        if (t < s) red[t] += red[t + s];
        __syncthreads();
    }
    if (t == 0) atomicAdd(loss, red[0] * (1.0f / ((float)Bb * (float)Dd)));
}

extern "C" void kernel_launch(void* const* d_in, const int* in_sizes, int n_in,
                              void* d_out, int out_size, void* d_ws, size_t ws_size,
                              hipStream_t stream) {
    (void)in_sizes; (void)n_in; (void)out_size; (void)ws_size;
    const float* x     = (const float*)d_in[0];
    const float* W_enc = (const float*)d_in[1];
    const float* b_enc = (const float*)d_in[2];
    const float* W_dec = (const float*)d_in[3];
    const float* b_dec = (const float*)d_in[4];

    // d_out: [x_hat | z_topk | z_pre | topk_idx(float) | loss]
    float* out     = (float*)d_out;
    float* x_hat   = out;
    float* z_topk  = x_hat + (size_t)Bb * Dd;
    float* z_pre   = z_topk + (size_t)Bb * Nn;
    float* tk_idx  = z_pre + (size_t)Bb * Nn;
    float* loss    = tk_idx + (size_t)Bb * Kk;

    // workspace: [xc bf16 (16MB) | Wb bf16 (128MB) | idx (1MB) | val (1MB)]
    char* ws = (char*)d_ws;
    __bf16* xc  = (__bf16*)ws;
    __bf16* Wb  = (__bf16*)(ws + (size_t)Bb * Dd * 2);
    int*    wsi = (int*)  (ws + (size_t)Bb * Dd * 2 + (size_t)Nn * Dd * 2);
    float*  wsv = (float*)((char*)wsi + (size_t)Bb * Kk * 4);

    center_cast_kernel<<<(Bb * Dd) / 256, 256, 0, stream>>>(x, b_dec, xc, loss);
    castw_kernel<<<(int)(((size_t)Nn * Dd / 4) / 256), 256, 0, stream>>>(W_enc, Wb);

    dim3 eg(Nn / BN, Bb / BM);   // (256, 32)
    encode_kernel<<<eg, 256, 0, stream>>>(xc, Wb, b_enc, z_pre);

    const size_t topk_lds = ((size_t)Nn + 256) * sizeof(unsigned);  // 132 KB < 320 KB WGP LDS
    topk_kernel<<<Bb, 256, topk_lds, stream>>>(z_pre, z_topk, tk_idx, wsi, wsv);

    decode_kernel<<<Bb, 256, 0, stream>>>(x, W_dec, b_dec, wsi, wsv, x_hat, loss);
}
